// GatedRNN_33535104648000
// MI455X (gfx1250) — compile-verified, matrix operand order checked by reading, not run on hardware
//
#include <hip/hip_runtime.h>
#include <hip/hip_bf16.h>

// ---------------------------------------------------------------------------
// Gated RNN (B=256, T=512, D=64, H=1024) for MI455X / gfx1250.
//
//  * 16 persistent blocks, one per 16-row batch tile (recurrence is
//    independent per batch row -> no grid-wide sync, only 4 s_barriers/step).
//  * All matmuls on v_wmma_f32_16x16x32_bf16 (bf16 in, fp32 accum).
//    Weights repacked once into bf16 B-fragment layout in d_ws (~2.3 MB),
//    L2-resident for the whole run (192 MB L2).
//  * Software-pipelined main GEMM: next kf's B-fragments (global) and
//    A-fragment (LDS) prefetched into a double buffer while the current
//    8 WMMAs execute (steady-state waits are loadcnt<=6, not 0).
//  * W3 out-GEMM K-split across all 8 waves (partials combined in LDS);
//    u-gate dot parallelized across 32 lanes.
// ---------------------------------------------------------------------------

typedef __attribute__((ext_vector_type(16))) __bf16 v16bf;
typedef __attribute__((ext_vector_type(8)))  float  v8f;

#define SEQ_D 64
#define HID   1024
#define BATCH 256
#define TIME  512

#define KF12  34            // K fragments for [W1|W2]: (64+1024)/32
#define NT12  64            // N tiles over H=1024
#define KF3   32            // K fragments for W3: 1024/32
#define NT3   4             // N tiles over D=64
#define FRAG_HALVES 512     // 32x16 bf16 elements per fragment (1 KB)

// A-matrix 16-bit 16x32 layout (ISA 7.12.2): lane = M + 16*hi,
// half hh in a lane covers K = khalf(hi, hh). B layout mirrors with M<->N.
__device__ __forceinline__ int khalf(int hi, int hh) {
    return (hh < 8) ? (hi * 8 + hh) : (16 + hi * 8 + (hh - 8));
}

// ---------------------------------------------------------------------------
// Repack W1|W2 and W3 (fp32, row-major [out,in]) into bf16 B-fragments.
// ---------------------------------------------------------------------------
__global__ void prep_frags(const float* __restrict__ W1,
                           const float* __restrict__ W2,
                           const float* __restrict__ W3,
                           __bf16* __restrict__ ws12,
                           __bf16* __restrict__ ws3) {
    long idx = (long)blockIdx.x * blockDim.x + threadIdx.x;
    const long N12 = (long)KF12 * NT12 * FRAG_HALVES;
    const long N3  = (long)KF3 * NT3 * FRAG_HALVES;
    if (idx < N12) {
        int f = (int)(idx >> 9), e = (int)(idx & 511);
        int kk = e >> 4, n = e & 15;
        int K = (f >> 6) * 32 + kk;            // frag index f = kf*64 + nt
        int N = (f & 63) * 16 + n;
        float v = (K < SEQ_D) ? W1[N * SEQ_D + K] : W2[N * HID + (K - SEQ_D)];
        int hi   = (kk >> 3) & 1;
        int half = (kk & 7) + ((kk >> 4) & 1) * 8;
        int lane = n + 16 * hi;
        ws12[(long)f * FRAG_HALVES + lane * 16 + half] = (__bf16)v;
    } else if (idx < N12 + N3) {
        long i = idx - N12;
        int f = (int)(i >> 9), e = (int)(i & 511);
        int kk = e >> 4, n = e & 15;
        int K = (f >> 2) * 32 + kk;            // frag index f = kf*4 + nt
        int N = (f & 3) * 16 + n;
        float v = W3[N * HID + K];
        int hi   = (kk >> 3) & 1;
        int half = (kk & 7) + ((kk >> 4) & 1) * 8;
        int lane = n + 16 * hi;
        ws3[(long)f * FRAG_HALVES + lane * 16 + half] = (__bf16)v;
    }
}

// ---------------------------------------------------------------------------
// Main persistent kernel: 16 blocks x 256 threads (8 wave32s).
// Wave w owns hidden columns [w*128, w*128+128) = 8 N-tiles of the pre GEMM.
// ---------------------------------------------------------------------------
__global__ __launch_bounds__(256, 1)
void gated_rnn_main(const float* __restrict__ x,
                    const float* __restrict__ b1, const float* __restrict__ b2,
                    const float* __restrict__ b3, const float* __restrict__ Wg,
                    const float* __restrict__ bg,
                    const __bf16* __restrict__ ws12,
                    const __bf16* __restrict__ ws3,
                    float* __restrict__ out) {
    __shared__ __align__(32) __bf16 hA[32 * FRAG_HALVES];   // 32 KB: h as A-fragments
    __shared__ float xbuf[2][16 * SEQ_D];                    // 8 KB: x double buffer
    __shared__ float outbuf2[2][16 * SEQ_D];                 // 8 KB: W3 K-split partials
    __shared__ float u_lds[16];
    __shared__ float upart[32];
    __shared__ float wg_lds[SEQ_D];
    __shared__ float b3_lds[SEQ_D];
    __shared__ float bg_lds;

    const int tid     = threadIdx.x;
    const int wave    = tid >> 5;        // wave32
    const int lane    = tid & 31;
    const int mlo     = lane & 15;
    const int chi     = lane >> 4;
    const int bx      = blockIdx.x;      // batch rows bx*16 .. bx*16+15
    const int nt0     = wave * 8;        // first N tile of this wave
    const int colbase = wave * 128;

    // fragment accessors ----------------------------------------------------
    const __bf16* wsl = ws12 + lane * 16;     // lane-resolved base
    const __bf16* w3l = ws3  + lane * 16;
    auto loadB12 = [&](int kf, v16bf* dst) {
        const __bf16* bp = wsl + ((long)kf * NT12 + nt0) * FRAG_HALVES;
        #pragma unroll
        for (int tt = 0; tt < 8; ++tt)
            dst[tt] = *(const v16bf*)(bp + tt * FRAG_HALVES);
    };
    auto ldsA = [&](int f) -> v16bf {
        return *(const v16bf*)&hA[f * FRAG_HALVES + lane * 16];
    };

    // ---- init -------------------------------------------------------------
    for (int i = tid; i < 32 * FRAG_HALVES; i += 256) hA[i] = (__bf16)0.0f;  // h0 = 0
    if (tid < 16) u_lds[tid] = 1.0f;                                          // u0 = 1
    if (tid < SEQ_D) { wg_lds[tid] = Wg[tid]; b3_lds[tid] = b3[tid]; }
    if (tid == 0) bg_lds = bg[0];
    {   // preload x[t=0] tile
        int row = tid >> 4, c4 = (tid & 15) * 4;
        float4 v = *(const float4*)&x[(((long)(bx * 16 + row)) * TIME + 0) * SEQ_D + c4];
        *(float4*)&xbuf[0][row * SEQ_D + c4] = v;
    }
    float bcomb[8];                       // b1+b2 for this thread's column (per tile)
    v8f   hold[8] = {};                   // fp32 h state, C-fragment layout
    #pragma unroll
    for (int tt = 0; tt < 8; ++tt) {
        int j = colbase + tt * 16 + mlo;
        bcomb[tt] = b1[j] + b2[j];
    }
    __syncthreads();

    // ---- time loop --------------------------------------------------------
    for (int t = 0; t < TIME; ++t) {
        const int cb = t & 1, nb = (t + 1) & 1;

        // (a) prefetch x[min(t+1, T-1)] tile -> xbuf[nb] (doubles as xn)
        {
            int tn  = (t + 1 < TIME) ? t + 1 : TIME - 1;
            int row = tid >> 4, c4 = (tid & 15) * 4;
            float4 v = *(const float4*)&x[(((long)(bx * 16 + row)) * TIME + tn) * SEQ_D + c4];
            *(float4*)&xbuf[nb][row * SEQ_D + c4] = v;
        }

        // (b) build xt A-fragments (2 frags of 16x32) from xbuf[cb]
        v16bf xa0, xa1;
        #pragma unroll
        for (int hh = 0; hh < 16; ++hh) {
            int k0 = khalf(chi, hh);
            xa0[hh] = (__bf16)xbuf[cb][mlo * SEQ_D + k0];
            xa1[hh] = (__bf16)xbuf[cb][mlo * SEQ_D + 32 + k0];
        }

        // (c) pre = [x | h] @ [W1 | W2]^T, software-pipelined.
        v8f acc[8] = {};
        {
            v16bf Bb[2][8];
            v16bf Ab[2];
            loadB12(0, Bb[0]);
            loadB12(1, Bb[1]);
            #pragma unroll
            for (int tt = 0; tt < 8; ++tt)             // kf = 0 : A = x lo
                acc[tt] = __builtin_amdgcn_wmma_f32_16x16x32_bf16(
                    false, xa0, false, Bb[0][tt], (short)0, acc[tt], false, false);
            loadB12(2, Bb[0]);                         // prefetch kf = 2
            Ab[0] = ldsA(0);
            #pragma unroll
            for (int tt = 0; tt < 8; ++tt)             // kf = 1 : A = x hi
                acc[tt] = __builtin_amdgcn_wmma_f32_16x16x32_bf16(
                    false, xa1, false, Bb[1][tt], (short)0, acc[tt], false, false);
            // kf = 2..33 : A = h fragments from LDS, B prefetched one ahead
            #pragma unroll 2
            for (int kf = 2; kf < KF12; ++kf) {
                const int cur = kf & 1, nxt = cur ^ 1;
                if (kf + 1 < KF12) {
                    loadB12(kf + 1, Bb[nxt]);
                    Ab[nxt] = ldsA(kf - 1);            // A frag for kf+1
                }
                #pragma unroll
                for (int tt = 0; tt < 8; ++tt)
                    acc[tt] = __builtin_amdgcn_wmma_f32_16x16x32_bf16(
                        false, Ab[cur], false, Bb[cur][tt], (short)0, acc[tt],
                        false, false);
            }
        }

        // (d) gate: h = u*tanh(pre+b) + (1-u)*h   (u from previous step)
        float uu[8];
        #pragma unroll
        for (int r = 0; r < 8; ++r) uu[r] = u_lds[r + 8 * chi];
        #pragma unroll
        for (int tt = 0; tt < 8; ++tt)
            #pragma unroll
            for (int r = 0; r < 8; ++r) {
                float pre = acc[tt][r] + bcomb[tt];
                hold[tt][r] = uu[r] * tanhf(pre) + (1.0f - uu[r]) * hold[tt][r];
            }

        __syncthreads();   // (e) everyone finished reading old hA / u_lds

        // write new h as bf16 A-fragments (inverse of khalf mapping)
        #pragma unroll
        for (int tt = 0; tt < 8; ++tt) {
            int j    = colbase + tt * 16 + mlo;   // hidden index = K of next GEMM
            int kf2  = j >> 5, kk = j & 31;
            int hi2  = (kk >> 3) & 1;
            int half = (kk & 7) + ((kk >> 4) & 1) * 8;
            #pragma unroll
            for (int r = 0; r < 8; ++r)
                hA[kf2 * FRAG_HALVES + (r + 8 * chi + 16 * hi2) * 16 + half] =
                    (__bf16)hold[tt][r];
        }

        __syncthreads();   // (f) new hA complete

        // (g) out = h @ W3^T, K-split across all 8 waves:
        //     wave w computes output tile (w&3), K range [(w>>2)*16, +16) kf.
        {
            const int otile = wave & 3;
            const int kf0   = (wave >> 2) * 16;
            v8f oacc = {};
            v16bf A2[2], B2[2];
            A2[0] = ldsA(kf0);
            B2[0] = *(const v16bf*)(w3l + ((long)kf0 * NT3 + otile) * FRAG_HALVES);
            #pragma unroll 2
            for (int k = 0; k < 16; ++k) {
                const int cur = k & 1, nxt = cur ^ 1;
                if (k + 1 < 16) {
                    A2[nxt] = ldsA(kf0 + k + 1);
                    B2[nxt] = *(const v16bf*)(w3l + ((long)(kf0 + k + 1) * NT3 + otile) * FRAG_HALVES);
                }
                oacc = __builtin_amdgcn_wmma_f32_16x16x32_bf16(
                    false, A2[cur], false, B2[cur], (short)0, oacc, false, false);
            }
            int n = otile * 16 + mlo;
            #pragma unroll
            for (int r = 0; r < 8; ++r)
                outbuf2[wave >> 2][(r + 8 * chi) * SEQ_D + n] = oacc[r];
        }

        __syncthreads();   // (h) both partial halves ready

        // gate: u = round(sigmoid(|xn - out| . Wg + bg)); 32 lanes of wave 7,
        // each lane handles one row x one 32-column half; combine on lanes 0-15.
        if (wave == 7) {
            int m = mlo, c0 = chi * 32;
            float z = 0.0f;
            #pragma unroll 4
            for (int c = c0; c < c0 + 32; ++c) {
                float o = outbuf2[0][m * SEQ_D + c] + outbuf2[1][m * SEQ_D + c] + b3_lds[c];
                z += fabsf(xbuf[nb][m * SEQ_D + c] - o) * wg_lds[c];
            }
            upart[lane] = z;                       // intra-wave LDS: in-order
            if (lane < 16) {
                float zz = bg_lds + upart[lane] + upart[lane + 16];
                float s  = 1.0f / (1.0f + __expf(-zz));
                u_lds[lane] = rintf(s);
            }
        }
        // combine partials + bias, coalesced store of out tile to global
        {
            int row = tid >> 4, c4 = (tid & 15) * 4;
            float4 v;
            v.x = outbuf2[0][row * SEQ_D + c4 + 0] + outbuf2[1][row * SEQ_D + c4 + 0] + b3_lds[c4 + 0];
            v.y = outbuf2[0][row * SEQ_D + c4 + 1] + outbuf2[1][row * SEQ_D + c4 + 1] + b3_lds[c4 + 1];
            v.z = outbuf2[0][row * SEQ_D + c4 + 2] + outbuf2[1][row * SEQ_D + c4 + 2] + b3_lds[c4 + 2];
            v.w = outbuf2[0][row * SEQ_D + c4 + 3] + outbuf2[1][row * SEQ_D + c4 + 3] + b3_lds[c4 + 3];
            *(float4*)&out[(((long)(bx * 16 + row)) * TIME + t) * SEQ_D + c4] = v;
        }

        __syncthreads();   // (i) u_lds ready for next step; outbuf2 reads done
    }
}

// ---------------------------------------------------------------------------
extern "C" void kernel_launch(void* const* d_in, const int* in_sizes, int n_in,
                              void* d_out, int out_size, void* d_ws, size_t ws_size,
                              hipStream_t stream) {
    const float* x  = (const float*)d_in[0];
    const float* W1 = (const float*)d_in[1];
    const float* b1 = (const float*)d_in[2];
    const float* W2 = (const float*)d_in[3];
    const float* b2 = (const float*)d_in[4];
    const float* W3 = (const float*)d_in[5];
    const float* b3 = (const float*)d_in[6];
    const float* Wg = (const float*)d_in[7];
    const float* bg = (const float*)d_in[8];

    __bf16* ws12 = (__bf16*)d_ws;
    __bf16* ws3  = (__bf16*)((char*)d_ws + (size_t)KF12 * NT12 * FRAG_HALVES * 2);

    long total  = (long)(KF12 * NT12 + KF3 * NT3) * FRAG_HALVES;
    int  blocks = (int)((total + 255) / 256);
    prep_frags<<<blocks, 256, 0, stream>>>(W1, W2, W3, ws12, ws3);
    gated_rnn_main<<<BATCH / 16, 256, 0, stream>>>(x, b1, b2, b3, Wg, bg,
                                                   ws12, ws3, (float*)d_out);
}
